// StandardAttention_61993557950732
// MI455X (gfx1250) — compile-verified, matrix operand order checked by reading, not run on hardware
//
#include <hip/hip_runtime.h>
#include <hip/hip_bf16.h>

// Problem constants (from reference)
#define BB 2
#define NN_ 2048
#define CC 1024
#define HH 16
#define DD 64
#define SCALE 0.125f          // 1/sqrt(64)
#define MASK_VAL -10000.0f

typedef __attribute__((ext_vector_type(16))) __bf16 v16bf;
typedef __attribute__((ext_vector_type(8)))  float  v8f;

// ---------------------------------------------------------------------------
// WMMA helpers (CDNA5 wave32, V_WMMA_F32_16X16X32_BF16)
// ---------------------------------------------------------------------------
__device__ __forceinline__ v8f wmma_bf16(v16bf a, v16bf b, v8f c) {
  return __builtin_amdgcn_wmma_f32_16x16x32_bf16(
      /*neg_a=*/false, a, /*neg_b=*/false, b,
      /*c_mod=*/(short)0, c, /*reuse_a=*/false, /*reuse_b=*/false);
}

// Load a 16x32 (MxK) bf16 fragment from a row-major [rows][stride] buffer.
// ISA layout (CDNA5 ISA, 16-bit A 16x32): lane = 16*half + m;
// VGPR j holds K = (j<4?0:16) + (j%4)*2 + 8*half + {0,1}.
// Same per-lane K-major mapping is used for B (rows == B columns, K-contig).
__device__ __forceinline__ v16bf load_frag(const __bf16* p, int rowBase,
                                           int stride, int kBase, int lane) {
  const int r = rowBase + (lane & 15);
  const int kh = kBase + ((lane >> 4) << 3);       // +8 K-offset for upper half
  const __bf16* row = p + (size_t)r * stride + kh;
  v16bf f;
#pragma unroll
  for (int j = 0; j < 8; ++j) {
    const int kb = ((j & 4) << 2) + ((j & 3) << 1);
    f[2 * j]     = row[kb];
    f[2 * j + 1] = row[kb + 1];
  }
  return f;
}

// ---------------------------------------------------------------------------
// Kernel 1: fused QKV projection  QKV[m,o] = sum_c X[m,c] * W[o,c]
//   X: [B*N, C] f32     W: [3C, C] f32 (torch Linear, y = x @ W^T)
// Writes bf16:  Qb,Kb as [B,H,N,D],  Vt as [B,H,D,N] (pre-transposed for P@V)
// Tile: 128(M) x 64(O), BK=32, 8 waves -> each wave 32x32 (4 WMMA tiles)
// ---------------------------------------------------------------------------
#define XSTR 34   // padded bf16 row stride (pairs stay 4B-aligned, banks skewed)

__global__ __launch_bounds__(256)
void qkv_gemm_kernel(const float* __restrict__ X, const float* __restrict__ W,
                     __bf16* __restrict__ Qb, __bf16* __restrict__ Kb,
                     __bf16* __restrict__ Vt) {
  __shared__ __bf16 Xs[128 * XSTR];
  __shared__ __bf16 Ws[64 * XSTR];

  const int tid  = threadIdx.x;
  const int lane = tid & 31;
  const int wave = tid >> 5;
  const int wm = (wave & 3) * 32;   // wave M offset within 128-tile
  const int wo = (wave >> 2) * 32;  // wave O offset within 64-tile
  const int mBase = blockIdx.x * 128;
  const int oBase = blockIdx.y * 64;

  v8f acc[2][2] = {};

  for (int k0 = 0; k0 < CC; k0 += 32) {
    // Stage X tile (128x32) and W tile (64x32) as bf16, coalesced f32 reads.
    for (int e = tid; e < 128 * 32; e += 256) {
      const int r = e >> 5, c = e & 31;
      Xs[r * XSTR + c] = (__bf16)X[(size_t)(mBase + r) * CC + k0 + c];
    }
    for (int e = tid; e < 64 * 32; e += 256) {
      const int r = e >> 5, c = e & 31;
      Ws[r * XSTR + c] = (__bf16)W[(size_t)(oBase + r) * CC + k0 + c];
    }
    __syncthreads();

    v16bf aF0 = load_frag(Xs, wm +  0, XSTR, 0, lane);
    v16bf aF1 = load_frag(Xs, wm + 16, XSTR, 0, lane);
    v16bf bF0 = load_frag(Ws, wo +  0, XSTR, 0, lane);
    v16bf bF1 = load_frag(Ws, wo + 16, XSTR, 0, lane);

    acc[0][0] = wmma_bf16(aF0, bF0, acc[0][0]);
    acc[0][1] = wmma_bf16(aF0, bF1, acc[0][1]);
    acc[1][0] = wmma_bf16(aF1, bF0, acc[1][0]);
    acc[1][1] = wmma_bf16(aF1, bF1, acc[1][1]);
    __syncthreads();
  }

  // Scatter-store into attention-friendly layouts.
  const int hh = lane >> 4, nn = lane & 15;
#pragma unroll
  for (int i = 0; i < 2; ++i)
#pragma unroll
    for (int j = 0; j < 2; ++j)
#pragma unroll
      for (int r = 0; r < 8; ++r) {
        const int m = mBase + wm + i * 16 + r + 8 * hh;
        const int o = oBase + wo + j * 16 + nn;
        const float v = acc[i][j][r];
        const int b = m >> 11, tok = m & (NN_ - 1);
        const int three = o >> 10, rem = o & 1023;
        const int h = rem >> 6, d = rem & 63;
        const size_t bh = (size_t)(b * HH + h);
        if (three == 0)
          Qb[(bh * NN_ + tok) * DD + d] = (__bf16)v;
        else if (three == 1)
          Kb[(bh * NN_ + tok) * DD + d] = (__bf16)v;
        else
          Vt[(bh * DD + d) * NN_ + tok] = (__bf16)v;   // transposed V
      }
}

// ---------------------------------------------------------------------------
// Kernel 2: attention for one (b, h, 16-row stripe).
// Entire 16x2048 f32 score stripe lives in LDS (CDNA5: 320KB/WGP) ->
// single-pass masked softmax with zero HBM round-trip; att_weights written
// once (mandatory output); P kept in LDS as bf16 for the P@V WMMA.
// ---------------------------------------------------------------------------
// dynamic LDS layout (bytes):
#define OFF_SF   0                         // float [16][2048]   131072
#define OFF_PB   131072                    // bf16  [16][2048]    65536
#define OFF_QS   196608                    // bf16  [16][64]       2048
#define OFF_OACC 198656                    // float [16][64]       4096
#define OFF_RED  202752                    // float [16][16]       1024
#define OFF_RMAX 203776                    // float [16]             64
#define OFF_RSUM 203840                    // float [16]             64
#define OFF_MCOL 203904                    // u8    [2048]         2048
#define OFF_MROW 205952                    // u8    [16]             16
#define SMEM_BYTES 206080

__global__ __launch_bounds__(256)
void attn_kernel(const __bf16* __restrict__ Qb, const __bf16* __restrict__ Kb,
                 const __bf16* __restrict__ Vt, const int* __restrict__ mask,
                 float* __restrict__ att_out, float* __restrict__ attw_out) {
  extern __shared__ char smem[];
  float*  Sf    = (float*)(smem + OFF_SF);
  __bf16* Pb    = (__bf16*)(smem + OFF_PB);
  __bf16* Qs    = (__bf16*)(smem + OFF_QS);
  float*  Oacc  = (float*)(smem + OFF_OACC);
  float*  red   = (float*)(smem + OFF_RED);
  float*  rmax  = (float*)(smem + OFF_RMAX);
  float*  rsum  = (float*)(smem + OFF_RSUM);
  unsigned char* mcol = (unsigned char*)(smem + OFF_MCOL);
  unsigned char* mrow = (unsigned char*)(smem + OFF_MROW);

  const int i0 = blockIdx.x * 16;     // row stripe
  const int h  = blockIdx.y;
  const int b  = blockIdx.z;
  const int tid = threadIdx.x, lane = tid & 31, wave = tid >> 5;
  const size_t bh = (size_t)(b * HH + h);

  const __bf16* Qg = Qb + (bh * NN_ + i0) * DD;
  const __bf16* Kg = Kb + bh * NN_ * DD;
  const __bf16* Vg = Vt + bh * DD * NN_;

  // Stage Q stripe, row/col padding masks.
  for (int e = tid; e < 16 * DD; e += 256) Qs[e] = Qg[e];
  for (int e = tid; e < NN_; e += 256)
    mcol[e] = (unsigned char)(mask[b * NN_ + e] != 0);
  if (tid < 16) mrow[tid] = (unsigned char)(mask[b * NN_ + i0 + tid] != 0);
  __syncthreads();

  // Persistent Q fragments (K = 0..31 and 32..63).
  const v16bf aq0 = load_frag(Qs, 0, DD, 0, lane);
  const v16bf aq1 = load_frag(Qs, 0, DD, 32, lane);

  // ---- S = scale * Q K^T with mask, stripe -> LDS ----
  const int hh = lane >> 4, nn = lane & 15;
  for (int jt = wave; jt < NN_ / 16; jt += 8) {
    const int j0 = jt * 16;
    if (j0 + 128 < NN_)
      __builtin_prefetch(Kg + (size_t)(j0 + 128) * DD, 0, 1);  // global_prefetch_b8
    v16bf bk0 = load_frag(Kg, j0, DD, 0, lane);
    v16bf bk1 = load_frag(Kg, j0, DD, 32, lane);
    v8f c = {};
    c = wmma_bf16(aq0, bk0, c);
    c = wmma_bf16(aq1, bk1, c);
    const int j = j0 + nn;
    const bool mj = (mcol[j] != 0);
#pragma unroll
    for (int r = 0; r < 8; ++r) {
      const int il = r + 8 * hh;
      float s = c[r] * SCALE;
      if (mj || mrow[il]) s = MASK_VAL;
      Sf[il * NN_ + j] = s;
    }
  }
  __syncthreads();

  // ---- softmax over the resident stripe: 16 groups x 16 threads ----
  const int row = tid >> 4, sub = tid & 15;
  {
    float m = -3.0e38f;
    for (int j = sub; j < NN_; j += 16) m = fmaxf(m, Sf[row * NN_ + j]);
    red[row * 16 + sub] = m;
  }
  __syncthreads();
  if (tid < 16) {
    float m = red[tid * 16];
    for (int s = 1; s < 16; ++s) m = fmaxf(m, red[tid * 16 + s]);
    rmax[tid] = m;
  }
  __syncthreads();
  {
    const float m = rmax[row];
    float acc = 0.f;
    for (int j = sub; j < NN_; j += 16) {
      const float e = __expf(Sf[row * NN_ + j] - m);
      Sf[row * NN_ + j] = e;
      acc += e;
    }
    red[row * 16 + sub] = acc;
  }
  __syncthreads();
  if (tid < 16) {
    float s = 0.f;
    for (int k = 0; k < 16; ++k) s += red[tid * 16 + k];
    rsum[tid] = s;
  }
  for (int e = tid; e < 16 * DD; e += 256) Oacc[e] = 0.f;  // zero P@V accum
  __syncthreads();
  {
    const float inv = 1.0f / rsum[row];
    float* wout = attw_out + (bh * NN_ + i0) * (size_t)NN_;
    for (int j = sub; j < NN_; j += 16) {
      const float p = Sf[row * NN_ + j] * inv;
      wout[(size_t)row * NN_ + j] = p;   // required output: att_weights
      Pb[row * NN_ + j] = (__bf16)p;     // bf16 copy for P@V WMMA
    }
  }
  __syncthreads();

  // ---- O = P @ V, split-K across 8 waves (256 K each), ds_add_f32 reduce ----
  v8f o[4] = {};
  for (int kc = 0; kc < 8; ++kc) {
    const int kb = wave * 256 + kc * 32;
    v16bf ap = load_frag(Pb, 0, NN_, kb, lane);
#pragma unroll
    for (int t = 0; t < 4; ++t) {
      v16bf bv = load_frag(Vg, t * 16, NN_, kb, lane);  // Vt rows are d, K-contig
      o[t] = wmma_bf16(ap, bv, o[t]);
    }
  }
#pragma unroll
  for (int t = 0; t < 4; ++t)
#pragma unroll
    for (int r = 0; r < 8; ++r)
      atomicAdd(&Oacc[(r + 8 * hh) * DD + t * 16 + nn], o[t][r]);
  __syncthreads();

  // att[b, n, h*D + d]
  for (int e = tid; e < 16 * DD; e += 256) {
    const int il = e / DD, d = e % DD;
    att_out[((size_t)(b * NN_ + i0 + il)) * CC + h * DD + d] = Oacc[e];
  }
}

// ---------------------------------------------------------------------------
extern "C" void kernel_launch(void* const* d_in, const int* in_sizes, int n_in,
                              void* d_out, int out_size, void* d_ws, size_t ws_size,
                              hipStream_t stream) {
  (void)in_sizes; (void)n_in; (void)out_size; (void)ws_size;
  const float* X    = (const float*)d_in[0];   // hidden_state [B,N,C]
  const float* W    = (const float*)d_in[1];   // W [3C, C]
  const int*   mask = (const int*)d_in[2];     // attention_mask [B,N]

  float* att  = (float*)d_out;                          // [B,N,C]
  float* attw = att + (size_t)BB * NN_ * CC;            // [B,H,N,N]

  __bf16* Qb = (__bf16*)d_ws;                           // [B,H,N,D] bf16
  __bf16* Kb = Qb + (size_t)BB * HH * NN_ * DD;
  __bf16* Vt = Kb + (size_t)BB * HH * NN_ * DD;         // [B,H,D,N] bf16

  dim3 g1((BB * NN_) / 128, (3 * CC) / 64);
  qkv_gemm_kernel<<<g1, 256, 0, stream>>>(X, W, Qb, Kb, Vt);

  // Raise dynamic-LDS cap (CDNA5 WGP has 320KB); not a stream op, capture-safe.
  hipFuncSetAttribute((const void*)attn_kernel,
                      hipFuncAttributeMaxDynamicSharedMemorySize, SMEM_BYTES);

  dim3 g2(NN_ / 16, HH, BB);
  attn_kernel<<<g2, 256, SMEM_BYTES, stream>>>(Qb, Kb, Vt, mask, att, attw);
}
